// LSTM_75299366634013
// MI455X (gfx1250) — matrix operand order repaired and checked by reading, not compiled, not run on hardware
//
#include <hip/hip_runtime.h>
#include <hip/hip_bf16.h>
#include <stdint.h>

typedef __attribute__((ext_vector_type(16))) __bf16 v16bf;
typedef __attribute__((ext_vector_type(8)))  __bf16 v8bf;
typedef __attribute__((ext_vector_type(8)))  float  v8f;

// pointer types for the async global->LDS builtin (b128 = int4 elements)
typedef int v4i_ __attribute__((vector_size(16)));
typedef __attribute__((address_space(1))) v4i_* g128p;
typedef __attribute__((address_space(3))) v4i_* l128p;

#define B_  64
#define S_  2048
#define F_  512
#define H_  512
#define NG_ 2048   /* 4*H */
#define KD_ 512

// ---------------- workspace layout (bytes) ----------------
#define XS_BYTES   ((unsigned long long)S_*4ull*B_*H_*4ull)      /* 1 GiB f32 gate pre-acts  */
#define W_BYTES    ((unsigned long long)NG_*KD_*2ull)            /* 2 MiB bf16 each          */
#define XS_OFF     0ull
#define WXT_OFF    (XS_OFF + XS_BYTES)
#define WHT_OFF    (WXT_OFF + W_BYTES)
#define BIAS_OFF   (WHT_OFF + W_BYTES)                           /* 8 KiB f32 folded biases  */
#define HBUF_OFF   (BIAS_OFF + 8192ull)                          /* 2 x 64x512 bf16          */
#define HBUF_BYTES (2ull*B_*H_*2ull)
#define CNT_OFF    (HBUF_OFF + HBUF_BYTES)
#define WS_NEED    (CNT_OFF + 256ull)

// dynamic LDS layout for the recurrent kernel
#define LDS_WL_BYTES  (128*KD_*2)        /* 128 KiB: W_h slice [128 gate cols][512 k] bf16 */
#define LDS_HA_BYTES  (16*KD_*2)         /* 16 KiB : h tile [16 m][512 k] bf16             */
#define LDS_Z_BYTES   (4*16*32*4)        /* 8 KiB  : gate pre-acts                         */
#define LDS_C_BYTES   (16*32*4)          /* 2 KiB  : persistent cell state                 */
#define LDS_TOTAL     (LDS_WL_BYTES + LDS_HA_BYTES + LDS_Z_BYTES + LDS_C_BYTES)

#if __has_builtin(__builtin_amdgcn_global_load_async_to_lds_b128)
#define ASYNC_LDS 1
#else
#define ASYNC_LDS 0
#endif

// Copy 16 bytes global -> LDS. On gfx1250 use the async DMA path
// (GLOBAL_LOAD_ASYNC_TO_LDS_B128, tracked by ASYNCcnt): no VGPR round-trip.
__device__ __forceinline__ void g2l_b128(void* lds_dst, const void* gsrc) {
#if ASYNC_LDS
  __builtin_amdgcn_global_load_async_to_lds_b128(
      (g128p)(uintptr_t)gsrc,
      (l128p)(unsigned)(uintptr_t)lds_dst,
      0, 0);
#else
  *(uint4*)lds_dst = *(const uint4*)gsrc;
#endif
}

__device__ __forceinline__ void wait_async_zero() {
#if ASYNC_LDS
#if __has_builtin(__builtin_amdgcn_s_wait_asynccnt)
  __builtin_amdgcn_s_wait_asynccnt(0);
#else
  asm volatile("s_wait_asynccnt 0x0" ::: "memory");
#endif
#endif
}

__device__ __forceinline__ float fast_sigmoid(float x) {
  return 1.0f / (1.0f + __expf(-x));
}
__device__ __forceinline__ float fast_tanh(float x) {
  float e = __expf(-2.0f * fabsf(x));
  float r = (1.0f - e) / (1.0f + e);
  return copysignf(r, x);
}

// ---------------------------------------------------------------------------
// Prep: build bf16 transposed weight matrices Wxt/Wht [N=2048][K=512],
// folded biases, zero h0 (bf16) and the group barrier counters.
// ---------------------------------------------------------------------------
__global__ __launch_bounds__(256) void lstm_prep(
    const float* __restrict__ wii, const float* __restrict__ whi,
    const float* __restrict__ wif, const float* __restrict__ whf,
    const float* __restrict__ wig, const float* __restrict__ whg,
    const float* __restrict__ wio, const float* __restrict__ who,
    const float* __restrict__ bii, const float* __restrict__ bhi,
    const float* __restrict__ bif, const float* __restrict__ bhf,
    const float* __restrict__ big, const float* __restrict__ bhg,
    const float* __restrict__ bio, const float* __restrict__ bho,
    __bf16* __restrict__ Wxt, __bf16* __restrict__ Wht,
    float* __restrict__ biasx, __bf16* __restrict__ hbuf,
    unsigned* __restrict__ cnt)
{
  const unsigned long long NW = (unsigned long long)NG_ * KD_;  // 1,048,576
  unsigned long long gid = (unsigned long long)blockIdx.x * 256u + threadIdx.x;

  if (gid < NW) {                       // Wxt[n][k] = w_x_gate[k][h]
    int n = (int)(gid >> 9), k = (int)(gid & 511);
    int g = n >> 9, h = n & 511;
    const float* w = (g == 0) ? wii : (g == 1) ? wif : (g == 2) ? wig : wio;
    Wxt[gid] = (__bf16)w[(size_t)k * H_ + h];
  } else if (gid < 2ull * NW) {         // Wht[n][k] = w_h_gate[k][h]
    unsigned long long id = gid - NW;
    int n = (int)(id >> 9), k = (int)(id & 511);
    int g = n >> 9, h = n & 511;
    const float* w = (g == 0) ? whi : (g == 1) ? whf : (g == 2) ? whg : who;
    Wht[id] = (__bf16)w[(size_t)k * H_ + h];
  } else if (gid < 2ull * NW + 2ull * B_ * H_) {   // zero both h buffers
    hbuf[gid - 2ull * NW] = (__bf16)0.0f;
  } else if (gid < 2ull * NW + 2ull * B_ * H_ + NG_) {
    int n = (int)(gid - (2ull * NW + 2ull * B_ * H_));
    int g = n >> 9, h = n & 511;
    const float* bi = (g == 0) ? bii : (g == 1) ? bif : (g == 2) ? big : bio;
    const float* bh = (g == 0) ? bhi : (g == 1) ? bhf : (g == 2) ? bhg : bho;
    biasx[n] = bi[h] + bh[h];
  } else if (gid < 2ull * NW + 2ull * B_ * H_ + NG_ + 4ull) {
    cnt[gid - (2ull * NW + 2ull * B_ * H_ + NG_)] = 0u;
  }
}

// ---------------------------------------------------------------------------
// Phase 1: xs[s][g][b][h] = (x @ Wx)[b*S+s][g*512+h] + (b_i+b_h)
// One big bf16 WMMA GEMM: M=131072, N=2048, K=512.
// WG = 256 threads = 8 waves laid out 4(M) x 2(N); each wave computes 16x128.
// ---------------------------------------------------------------------------
__global__ __launch_bounds__(256) void lstm_xproj(
    const float* __restrict__ x, const __bf16* __restrict__ Wxt,
    const float* __restrict__ biasx, float* __restrict__ xs)
{
  const int tid   = threadIdx.x;
  const int lane  = tid & 31;
  const int wave  = tid >> 5;
  const int wm    = wave & 3;
  const int wn    = wave >> 2;
  const int m0    = blockIdx.x * 64 + wm * 16;
  const int n0    = blockIdx.y * 256 + wn * 128;
  const int mrow  = lane & 15;   // A row / B column within tile
  const int khalf = lane >> 4;   // which K half-group this lane holds

  const float* ap = x + (size_t)(m0 + mrow) * F_;

  v8f acc[8] = {};

  #pragma unroll 4
  for (int kb = 0; kb < KD_; kb += 32) {
    // A fragment (16x32 bf16): lane holds K = kb+kh*16..+15
    v16bf a;
    #pragma unroll
    for (int i = 0; i < 8; ++i) {
      a[i]     = (__bf16)ap[kb + khalf * 16 + i];
      a[i + 8] = (__bf16)ap[kb + khalf * 16 + 8 + i];
    }
    __builtin_prefetch(ap + kb + 64, 0, 3);   // global_prefetch_b8 (near scope)
    #pragma unroll
    for (int t = 0; t < 8; ++t) {
      // B fragment (32x16 bf16) from transposed weights: 16 contiguous halves
      const v16bf b = *(const v16bf*)(Wxt + (size_t)(n0 + t * 16 + mrow) * KD_
                                          + kb + khalf * 16);
      acc[t] = __builtin_amdgcn_wmma_f32_16x16x32_bf16(
          false, a, false, b, (short)0, acc[t], false, false);
    }
  }

  // epilogue: add folded bias, scatter to xs[s][g][b][h]
  #pragma unroll
  for (int t = 0; t < 8; ++t) {
    const int n = n0 + t * 16 + mrow;
    const int g = n >> 9, h = n & 511;
    const float bb = biasx[n];
    #pragma unroll
    for (int r = 0; r < 8; ++r) {
      const int m = m0 + r + 8 * khalf;     // m = b*S + s
      const int s = m & (S_ - 1);
      const int bidx = m >> 11;
      xs[(((size_t)s * 4 + g) * B_ + bidx) * H_ + h] = acc[t][r] + bb;
    }
  }
}

// ---------------------------------------------------------------------------
// Phase 2: recurrence. 4 independent batch groups x 16 WGs.
// WG owns batch rows [16g,16g+16) and h-cols [32ch,32ch+32) -> 128 gate cols.
// W_h slice lives in LDS for the whole kernel (async-DMA'd in once); per step
// the group's h tile is async-DMA'd into LDS, 16 chained bf16 WMMAs per wave,
// h exchanged via L2 (bf16), synced with a monotonic atomic barrier.
// ---------------------------------------------------------------------------
__global__ __launch_bounds__(256) void lstm_recur(
    const float* __restrict__ xs, const __bf16* __restrict__ Wht,
    __bf16* __restrict__ hbuf, unsigned* __restrict__ cnt,
    float* __restrict__ out)
{
  extern __shared__ char smem[];
  __bf16* Wl = (__bf16*)smem;                                   // [128][512]
  __bf16* hA = (__bf16*)(smem + LDS_WL_BYTES);                  // [16][512]
  float*  Z  = (float*)(smem + LDS_WL_BYTES + LDS_HA_BYTES);    // [4][16][32]
  float*  Cs = (float*)(smem + LDS_WL_BYTES + LDS_HA_BYTES + LDS_Z_BYTES); // [16][32]

  const int tid   = threadIdx.x;
  const int lane  = tid & 31;
  const int wave  = tid >> 5;
  const int g     = wave >> 1;           // gate 0..3 (i,f,g,o)
  const int nsub  = (wave & 1) * 16;     // N sub-tile within 32 h-cols
  const int group = blockIdx.x >> 4;     // batch group 0..3
  const int ch    = blockIdx.x & 15;     // h-column chunk 0..15
  const int mb0   = group * 16;
  const int j0    = ch * 32;
  const int mrow  = lane & 15;
  const int khalf = lane >> 4;
  const int lr    = g * 32 + nsub + mrow; // local weight row (gate col)

  // stage the [512 x 128] W_h slice into LDS via async global->LDS DMA
  #pragma unroll
  for (int i = 0; i < 32; ++i) {
    int idx = tid + i * 256;             // 0..8191 uint4s
    int r   = idx >> 6;                  // 0..127
    int c16 = idx & 63;
    int n   = (r >> 5) * 512 + j0 + (r & 31);
    g2l_b128(((uint4*)Wl) + idx, Wht + (size_t)n * KD_ + c16 * 8);
  }
  Cs[tid] = 0.0f;
  Cs[tid + 256] = 0.0f;
  wait_async_zero();
  __syncthreads();

  const __bf16* wrow  = Wl + (size_t)lr * KD_ + khalf * 16;
  const __bf16* hrow0 = hA + (size_t)mrow * KD_;

  for (int t = 0; t < S_; ++t) {
    const int curb = t & 1;
    const __bf16* hb = hbuf + curb * (B_ * H_);

    // stage h(t) tile [16 x 512] bf16 into LDS via async DMA
    #pragma unroll
    for (int i = 0; i < 4; ++i) {
      int idx = tid + i * 256;           // 0..1023 uint4s
      int m   = idx >> 6;
      int c16 = idx & 63;
      g2l_b128(((uint4*)hA) + idx, hb + (size_t)(mb0 + m) * H_ + c16 * 8);
    }
    wait_async_zero();
    __syncthreads();

    // init accumulator from precomputed input projection (includes biases)
    v8f acc;
    {
      const float* xp = xs + (((size_t)t * 4 + g) * B_ + mb0) * H_
                           + j0 + nsub + mrow;
      #pragma unroll
      for (int r = 0; r < 8; ++r) acc[r] = xp[(size_t)(r + 8 * khalf) * H_];
    }

    // h(t) @ Wh slice : 16 chained bf16 WMMAs
    #pragma unroll
    for (int kb = 0; kb < KD_; kb += 32) {
      v16bf a;
      const v8bf a0 = *(const v8bf*)(hrow0 + kb + khalf * 16);
      const v8bf a1 = *(const v8bf*)(hrow0 + kb + khalf * 16 + 8);
      #pragma unroll
      for (int i = 0; i < 8; ++i) { a[i] = a0[i]; a[i + 8] = a1[i]; }
      const v16bf b = *(const v16bf*)(wrow + kb);
      acc = __builtin_amdgcn_wmma_f32_16x16x32_bf16(
          false, a, false, b, (short)0, acc, false, false);
    }

    // publish gate tile to LDS
    #pragma unroll
    for (int r = 0; r < 8; ++r)
      Z[((g * 16) + r + 8 * khalf) * 32 + nsub + mrow] = acc[r];
    __syncthreads();

    // elementwise LSTM cell (512 elements, 2 per thread)
    __bf16* hn_out = hbuf + (curb ^ 1) * (B_ * H_);
    #pragma unroll
    for (int e = 0; e < 2; ++e) {
      int idx = tid + e * 256;
      int m = idx >> 5;
      int j = idx & 31;
      float zi = Z[(0 * 16 + m) * 32 + j];
      float zf = Z[(1 * 16 + m) * 32 + j];
      float zg = Z[(2 * 16 + m) * 32 + j];
      float zo = Z[(3 * 16 + m) * 32 + j];
      float ig = fast_sigmoid(zi);
      float fg = fast_sigmoid(zf);
      float gg = fast_tanh(zg);
      float og = fast_sigmoid(zo);
      float cn = fg * Cs[idx] + ig * gg;
      Cs[idx] = cn;
      float hn = og * fast_tanh(cn);
      hn_out[(size_t)(mb0 + m) * H_ + j0 + j] = (__bf16)hn;
      if (t == S_ - 1) {
        out[(size_t)(mb0 + m) * H_ + j0 + j] = hn;
        out[(size_t)B_ * H_ + (size_t)(mb0 + m) * H_ + j0 + j] = cn;
      }
    }

    // cross-WG barrier for this batch group (monotonic counter)
    __threadfence();
    __syncthreads();
    if (tid == 0) {
      __hip_atomic_fetch_add(&cnt[group], 1u, __ATOMIC_RELEASE,
                             __HIP_MEMORY_SCOPE_AGENT);
      const unsigned target = 16u * (unsigned)(t + 1);
      while (__hip_atomic_load(&cnt[group], __ATOMIC_ACQUIRE,
                               __HIP_MEMORY_SCOPE_AGENT) < target) {
        __builtin_amdgcn_s_sleep(2);
      }
    }
    __syncthreads();
    __builtin_amdgcn_fence(__ATOMIC_ACQUIRE, "agent"); // invalidate stale WGP$ lines
  }
}

// ---------------------------------------------------------------------------
extern "C" void kernel_launch(void* const* d_in, const int* in_sizes, int n_in,
                              void* d_out, int out_size, void* d_ws, size_t ws_size,
                              hipStream_t stream) {
  (void)in_sizes; (void)n_in; (void)out_size;
  if (ws_size < (size_t)WS_NEED) return;

  const float* x   = (const float*)d_in[0];
  const float* wii = (const float*)d_in[1];
  const float* bii = (const float*)d_in[2];
  const float* whi = (const float*)d_in[3];
  const float* bhi = (const float*)d_in[4];
  const float* wif = (const float*)d_in[5];
  const float* bif = (const float*)d_in[6];
  const float* whf = (const float*)d_in[7];
  const float* bhf = (const float*)d_in[8];
  const float* wig = (const float*)d_in[9];
  const float* big = (const float*)d_in[10];
  const float* whg = (const float*)d_in[11];
  const float* bhg = (const float*)d_in[12];
  const float* wio = (const float*)d_in[13];
  const float* bio = (const float*)d_in[14];
  const float* who = (const float*)d_in[15];
  const float* bho = (const float*)d_in[16];

  char* ws = (char*)d_ws;
  float*    xs    = (float*)(ws + XS_OFF);
  __bf16*   Wxt   = (__bf16*)(ws + WXT_OFF);
  __bf16*   Wht   = (__bf16*)(ws + WHT_OFF);
  float*    biasx = (float*)(ws + BIAS_OFF);
  __bf16*   hbuf  = (__bf16*)(ws + HBUF_OFF);
  unsigned* cnt   = (unsigned*)(ws + CNT_OFF);
  float*    out   = (float*)d_out;

  // prep: weights->bf16 transposed, folded biases, zero h0 + counters
  {
    unsigned long long total = 2ull * NG_ * KD_ + 2ull * B_ * H_ + NG_ + 4ull;
    unsigned blocks = (unsigned)((total + 255ull) / 256ull);
    lstm_prep<<<blocks, 256, 0, stream>>>(
        wii, whi, wif, whf, wig, whg, wio, who,
        bii, bhi, bif, bhf, big, bhg, bio, bho,
        Wxt, Wht, biasx, hbuf, cnt);
  }

  // phase 1: big input-projection GEMM
  lstm_xproj<<<dim3((B_ * S_) / 64, NG_ / 256), 256, 0, stream>>>(
      x, Wxt, biasx, xs);

  // phase 2: recurrence (64 persistent WGs, 154 KiB dynamic LDS each)
  (void)hipFuncSetAttribute((const void*)lstm_recur,
                            hipFuncAttributeMaxDynamicSharedMemorySize, LDS_TOTAL);
  lstm_recur<<<64, 256, LDS_TOTAL, stream>>>(xs, Wht, hbuf, cnt, out);
}
